// OptimizedAttention_59588376264819
// MI455X (gfx1250) — matrix operand order repaired and checked by reading, not compiled, not run on hardware
//
#include <hip/hip_runtime.h>
#include <hip/hip_bf16.h>

typedef __bf16 bf16;
typedef __attribute__((ext_vector_type(16))) __bf16 v16bf;
typedef __attribute__((ext_vector_type(8)))  float   v8f;
typedef __attribute__((ext_vector_type(4)))  float   f32x4;
typedef __attribute__((ext_vector_type(4)))  unsigned int u32x4;
typedef __attribute__((ext_vector_type(8)))  int     i32x8;
typedef __attribute__((ext_vector_type(4)))  int     i32x4;

#define B_  4
#define T_  2048
#define C_  2048
#define H_  16
#define D_  128
#define M_  (B_*T_)   // 8192

union FragBF { v16bf v; f32x4 h[2]; };

__device__ inline v16bf ld_frag(const unsigned char* p0, const unsigned char* p1) {
  FragBF f;
  f.h[0] = *(const f32x4*)p0;
  f.h[1] = *(const f32x4*)p1;
  return f.v;
}

// ---- Tensor Data Mover: 2D tile (bf16) global -> LDS with row padding ----
// pad_interval code 3 => every 16 dwords (64B row), pad_amount code 3 => 4 dwords (16B)
__device__ inline void tdm_load_2d(unsigned lds_off, const void* gaddr,
                                   unsigned tensor_d0, unsigned tensor_d1,
                                   unsigned tile_d0, unsigned tile_d1,
                                   unsigned stride_d0) {
  const unsigned long long ga = (unsigned long long)gaddr;
  u32x4 g0;
  g0[0] = 1u;                                            // count=1 (valid), user mode
  g0[1] = lds_off;                                       // LDS byte address
  g0[2] = (unsigned)(ga & 0xffffffffu);                  // global_addr[31:0]
  g0[3] = (unsigned)((ga >> 32) & 0x01ffffffu)           // global_addr[56:32]
        | 0x80000000u;                                   // type=2 ("image")
  i32x8 g1;
  g1[0] = (int)((1u << 16)        // data_size = 1 (2 bytes)
              | (1u << 20)        // pad_enable
              | (3u << 22)        // pad_interval: 16 dwords
              | (3u << 25));      // pad_amount: 4 dwords
  g1[1] = (int)(tensor_d0 << 16);                        // tensor_dim0[15:0]
  g1[2] = (int)((tensor_d0 >> 16) | (tensor_d1 << 16));  // dim0 hi | dim1 lo
  g1[3] = (int)((tensor_d1 >> 16) | (tile_d0 << 16));    // dim1 hi | tile_dim0
  g1[4] = (int)(tile_d1 & 0xffffu);                      // tile_dim1, tile_dim2=0
  g1[5] = (int)stride_d0;                                // tensor_dim0_stride lo
  g1[6] = 0;                                             // stride hi | dim1_stride lo
  g1[7] = 0;
  const i32x4 gz4 = {0, 0, 0, 0};                        // groups 2/3: dims 2..4 unused
  const i32x8 gz8 = {0, 0, 0, 0, 0, 0, 0, 0};
  __builtin_amdgcn_tensor_load_to_lds(g0, g1, gz4, gz4, gz8, 0);
}

// ---------------- LayerNorm -> bf16 ----------------
__global__ __launch_bounds__(256) void ln_kernel(const float* __restrict__ x,
                                                 const float* __restrict__ g,
                                                 const float* __restrict__ b,
                                                 bf16* __restrict__ nx) {
  const int row = blockIdx.x;
  const float* xr = x + (size_t)row * C_;
  float v[8];
  float s = 0.f, s2 = 0.f;
#pragma unroll
  for (int i = 0; i < 8; ++i) {
    v[i] = xr[threadIdx.x + i * 256];
    s += v[i]; s2 += v[i] * v[i];
  }
#pragma unroll
  for (int m = 1; m < 32; m <<= 1) {
    s  += __shfl_xor(s,  m, 32);
    s2 += __shfl_xor(s2, m, 32);
  }
  __shared__ float red0[8], red1[8];
  const int wave = threadIdx.x >> 5;
  if ((threadIdx.x & 31) == 0) { red0[wave] = s; red1[wave] = s2; }
  __syncthreads();
  float ts = 0.f, ts2 = 0.f;
#pragma unroll
  for (int w = 0; w < 8; ++w) { ts += red0[w]; ts2 += red1[w]; }
  const float mu  = ts * (1.f / C_);
  const float var = ts2 * (1.f / C_) - mu * mu;
  const float inv = rsqrtf(var + 1e-5f);
  bf16* o = nx + (size_t)row * C_;
#pragma unroll
  for (int i = 0; i < 8; ++i) {
    const int c = threadIdx.x + i * 256;
    o[c] = (bf16)((v[i] - mu) * inv * g[c] + b[c]);
  }
}

// ---------------- Weight transpose fp32 [K][N] -> bf16 [N][K] ----------------
__global__ __launch_bounds__(256) void wtrans_kernel(const float* __restrict__ W,
                                                     bf16* __restrict__ Wt) {
  __shared__ bf16 tile[32][33];
  const int k0 = blockIdx.y * 32;
  const int n0 = blockIdx.x * 32;
  const int tx = threadIdx.x & 31;
  const int ty = threadIdx.x >> 5;
#pragma unroll
  for (int r = 0; r < 4; ++r) {
    const int k = ty + r * 8;
    tile[k][tx] = (bf16)W[(size_t)(k0 + k) * C_ + n0 + tx];
  }
  __syncthreads();
#pragma unroll
  for (int r = 0; r < 4; ++r) {
    const int n = ty + r * 8;
    Wt[(size_t)(n0 + n) * C_ + k0 + tx] = tile[tx][n];
  }
}

// ---------------- WMMA GEMM with TDM double-buffered staging ----------------
// mode 0: bf16 out scattered to [b][h][t][d]       (q, k)
// mode 1: bf16 out scattered to [b][h][d][t]       (v transposed)
// mode 2: fp32 out[m][n] = acc + bias + xres[m][n] (output proj + residual)
__global__ __launch_bounds__(256) void gemm_kernel(const bf16* __restrict__ A,
                                                   const bf16* __restrict__ Wt,
                                                   const float* __restrict__ bias,
                                                   void* __restrict__ outp,
                                                   const float* __restrict__ xres,
                                                   int mode) {
  constexpr int AP = 80;           // 32 bf16 (64B) + 16B TDM pad
  constexpr int ABYTES = 128 * AP; // 10240
  constexpr int BBYTES = 64 * AP;  // 5120
  __shared__ __align__(16) unsigned char sA[2 * ABYTES];
  __shared__ __align__(16) unsigned char sB[2 * BBYTES];
  const int tid  = threadIdx.x;
  const int lane = tid & 31, wave = tid >> 5;
  const int wm = wave & 3, wn = wave >> 2;
  const int lr = lane & 15, hi = lane >> 4;
  const int n0 = blockIdx.x * 64;
  const int m0 = blockIdx.y * 128;
  const unsigned ldsA = (unsigned)(unsigned long long)(void*)sA;
  const unsigned ldsB = (unsigned)(unsigned long long)(void*)sB;
  v8f acc[2][2] = {};

  // prologue: DMA tile 0 into buffer 0
  if (wave == 0) {
    tdm_load_2d(ldsA, A  + (size_t)m0 * C_, C_, M_, 32, 128, C_);
    tdm_load_2d(ldsB, Wt + (size_t)n0 * C_, C_, C_, 32, 64,  C_);
    __builtin_amdgcn_s_wait_tensorcnt(0);
  }
  __syncthreads();

  for (int kb = 0, it = 0; kb < C_; kb += 32, ++it) {
    const int cur = it & 1;
    const bool more = (kb + 32) < C_;
    // DMA next tile into the other buffer while we compute this one
    if (wave == 0 && more) {
      tdm_load_2d(ldsA + (cur ^ 1) * ABYTES, A  + (size_t)m0 * C_ + kb + 32,
                  C_, M_, 32, 128, C_);
      tdm_load_2d(ldsB + (cur ^ 1) * BBYTES, Wt + (size_t)n0 * C_ + kb + 32,
                  C_, C_, 32, 64, C_);
    }
    const unsigned char* cA = sA + cur * ABYTES;
    const unsigned char* cB = sB + cur * BBYTES;
    v16bf af[2], bfr[2];
#pragma unroll
    for (int mt = 0; mt < 2; ++mt) {
      const unsigned char* base = cA + (wm * 32 + mt * 16 + lr) * AP;
      af[mt] = ld_frag(base + hi * 16, base + 32 + hi * 16);       // A layout
    }
#pragma unroll
    for (int nt = 0; nt < 2; ++nt) {
      const unsigned char* base = cB + (wn * 32 + nt * 16 + lr) * AP;
      bfr[nt] = ld_frag(base + hi * 16, base + 32 + hi * 16);      // B layout
    }
#pragma unroll
    for (int mt = 0; mt < 2; ++mt)
#pragma unroll
      for (int nt = 0; nt < 2; ++nt)
        acc[mt][nt] = __builtin_amdgcn_wmma_f32_16x16x32_bf16(
            false, af[mt], false, bfr[nt], (short)0, acc[mt][nt], false, false);
    if (wave == 0 && more) __builtin_amdgcn_s_wait_tensorcnt(0);
    __syncthreads();
  }

#pragma unroll
  for (int mt = 0; mt < 2; ++mt) {
#pragma unroll
    for (int nt = 0; nt < 2; ++nt) {
      const int ng = n0 + wn * 32 + nt * 16 + lr;
      const float bv = bias[ng];
#pragma unroll
      for (int i = 0; i < 8; ++i) {
        const int mg = m0 + wm * 32 + mt * 16 + i + hi * 8;
        const float val = acc[mt][nt][i] + bv;
        if (mode == 2) {
          float* o = (float*)outp;
          o[(size_t)mg * C_ + ng] = val + xres[(size_t)mg * C_ + ng];
        } else {
          const int b = mg >> 11, t = mg & (T_ - 1);
          const int h = ng >> 7,  d = ng & (D_ - 1);
          bf16* o = (bf16*)outp;
          if (mode == 0)
            o[((size_t)(b * H_ + h) * T_ + t) * D_ + d] = (bf16)val;
          else
            o[((size_t)(b * H_ + h) * D_ + d) * T_ + t] = (bf16)val;
        }
      }
    }
  }
}

// ---------------- RoPE in-place on bf16 [b][h][t][d] ----------------
__global__ __launch_bounds__(256) void rope_kernel(bf16* __restrict__ buf) {
  const size_t idx = (size_t)blockIdx.x * 256 + threadIdx.x;  // B*H*T*64
  const int j = (int)(idx & 63);
  const size_t rowi = idx >> 6;
  const int t = (int)(rowi & (T_ - 1));
  bf16* p = buf + rowi * D_;
  const float invf = __expf(-((float)(2 * j) * (1.f / 128.f)) * 9.210340371976184f);
  const float ang = (float)t * invf;
  const float c = __cosf(ang), s = __sinf(ang);
  const float x1 = (float)p[j], x2 = (float)p[j + 64];
  p[j]      = (bf16)(x1 * c - x2 * s);
  p[j + 64] = (bf16)(x2 * c + x1 * s);
}

// ---------------- Flash attention (causal) ----------------
__global__ __launch_bounds__(128) void attn_kernel(const bf16* __restrict__ Q,
                                                   const bf16* __restrict__ K,
                                                   const bf16* __restrict__ VT,
                                                   bf16* __restrict__ ctx) {
  constexpr int KP = 272;  // 128 bf16 + 16B pad
  constexpr int VP = 80;   // 32 bf16 + 16B pad
  __shared__ __align__(16) unsigned char sK[32 * KP];     // K tile  [key][d]
  __shared__ __align__(16) unsigned char sV[128 * VP];    // V^T tile [d][key]
  __shared__ __align__(16) unsigned char sP[4 * 16 * VP]; // per-wave P staging
  const int tid  = threadIdx.x;
  const int lane = tid & 31, wave = tid >> 5;
  const int lr = lane & 15, hi = lane >> 4;
  const int bid  = blockIdx.x;
  const int qblk = bid & 31;             // T/64 blocks
  const int bh   = bid >> 5;
  const int b = bh >> 4, h = bh & 15;
  const int q0 = qblk * 64;
  const int qbase = q0 + wave * 16;
  const bf16* Qp = Q  + (size_t)(b * H_ + h) * T_ * D_;
  const bf16* Kp = K  + (size_t)(b * H_ + h) * T_ * D_;
  const bf16* Vp = VT + (size_t)(b * H_ + h) * D_ * T_;

  // Q fragments (A layout), direct from global: row = qbase + lr
  v16bf qf[4];
  {
    const unsigned char* qrow = (const unsigned char*)(Qp + (size_t)(qbase + lr) * D_);
#pragma unroll
    for (int c = 0; c < 4; ++c)
      qf[c] = ld_frag(qrow + c * 64 + hi * 16, qrow + c * 64 + 32 + hi * 16);
  }

  v8f o[8] = {};
  float mrow[8], lrow[8];
#pragma unroll
  for (int i = 0; i < 8; ++i) { mrow[i] = -1e30f; lrow[i] = 0.f; }
  const float scale = 0.08838834764831845f;  // 1/sqrt(128)

  for (int kb = 0; kb < q0 + 64; kb += 32) {
    // stage K tile (32 x 128 bf16)
#pragma unroll
    for (int p = 0; p < 4; ++p) {
      const int idx = p * 128 + tid;
      const int r = idx >> 4, cg = idx & 15;
      f32x4 d = *(const f32x4*)(Kp + (size_t)(kb + r) * D_ + cg * 8);
      *(f32x4*)(sK + r * KP + cg * 16) = d;
    }
    // stage V^T tile (128 x 32 bf16)
#pragma unroll
    for (int p = 0; p < 4; ++p) {
      const int idx = p * 128 + tid;
      const int r = idx >> 2, cg = idx & 3;
      f32x4 d = *(const f32x4*)(Vp + (size_t)r * T_ + kb + cg * 8);
      *(f32x4*)(sV + r * VP + cg * 16) = d;
    }
    __syncthreads();

    if (kb < qbase + 16) {
      v8f s0 = {}, s1 = {};
#pragma unroll
      for (int c = 0; c < 4; ++c) {
        const unsigned char* k0 = sK + lr * KP + c * 64;
        v16bf b0 = ld_frag(k0 + hi * 16, k0 + 32 + hi * 16);
        s0 = __builtin_amdgcn_wmma_f32_16x16x32_bf16(false, qf[c], false, b0,
                                                     (short)0, s0, false, false);
        const unsigned char* k1 = sK + (16 + lr) * KP + c * 64;
        v16bf b1 = ld_frag(k1 + hi * 16, k1 + 32 + hi * 16);
        s1 = __builtin_amdgcn_wmma_f32_16x16x32_bf16(false, qf[c], false, b1,
                                                     (short)0, s1, false, false);
      }
      // scale + causal mask (C layout: row = i + 8*hi, col = lr)
#pragma unroll
      for (int i = 0; i < 8; ++i) {
        const int qrow = qbase + i + hi * 8;
        float v0 = s0[i] * scale, v1 = s1[i] * scale;
        if (kb + lr      > qrow) v0 = -1e9f;
        if (kb + 16 + lr > qrow) v1 = -1e9f;
        s0[i] = v0; s1[i] = v1;
      }
      // online softmax over this 32-key block
      float fac[8];
#pragma unroll
      for (int i = 0; i < 8; ++i) {
        float mx = fmaxf(s0[i], s1[i]);
        mx = fmaxf(mx, __shfl_xor(mx, 1, 16));
        mx = fmaxf(mx, __shfl_xor(mx, 2, 16));
        mx = fmaxf(mx, __shfl_xor(mx, 4, 16));
        mx = fmaxf(mx, __shfl_xor(mx, 8, 16));
        const float mnew = fmaxf(mrow[i], mx);
        fac[i] = __expf(mrow[i] - mnew);
        mrow[i] = mnew;
        const float p0 = __expf(s0[i] - mnew);
        const float p1 = __expf(s1[i] - mnew);
        s0[i] = p0; s1[i] = p1;
        float rs = p0 + p1;
        rs += __shfl_xor(rs, 1, 16);
        rs += __shfl_xor(rs, 2, 16);
        rs += __shfl_xor(rs, 4, 16);
        rs += __shfl_xor(rs, 8, 16);
        lrow[i] = lrow[i] * fac[i] + rs;
      }
#pragma unroll
      for (int nt = 0; nt < 8; ++nt)
#pragma unroll
        for (int i = 0; i < 8; ++i) o[nt][i] *= fac[i];
      // transpose P (C layout -> A layout) through per-wave LDS
      unsigned char* pb = sP + wave * 16 * VP;
#pragma unroll
      for (int i = 0; i < 8; ++i) {
        const int r = i + hi * 8;
        *(bf16*)(pb + r * VP + lr * 2)        = (bf16)s0[i];
        *(bf16*)(pb + r * VP + (16 + lr) * 2) = (bf16)s1[i];
      }
      const unsigned char* prow = pb + lr * VP;
      v16bf pf = ld_frag(prow + hi * 16, prow + 32 + hi * 16);
      // O += P @ V  (B fragments from V^T rows)
#pragma unroll
      for (int nt = 0; nt < 8; ++nt) {
        const unsigned char* vb = sV + (nt * 16 + lr) * VP;
        v16bf vf = ld_frag(vb + hi * 16, vb + 32 + hi * 16);
        o[nt] = __builtin_amdgcn_wmma_f32_16x16x32_bf16(false, pf, false, vf,
                                                        (short)0, o[nt], false, false);
      }
    }
    __syncthreads();
  }

  // finalize: divide by l, write ctx [b*T+t][h*D+d] bf16
#pragma unroll
  for (int i = 0; i < 8; ++i) {
    const float invl = 1.f / lrow[i];
    const int t = qbase + i + hi * 8;
    const size_t rowo = (size_t)(b * T_ + t) * C_ + h * D_;
#pragma unroll
    for (int nt = 0; nt < 8; ++nt)
      ctx[rowo + nt * 16 + lr] = (bf16)(o[nt][i] * invl);
  }
}

extern "C" void kernel_launch(void* const* d_in, const int* in_sizes, int n_in,
                              void* d_out, int out_size, void* d_ws, size_t ws_size,
                              hipStream_t stream) {
  (void)in_sizes; (void)n_in; (void)out_size; (void)ws_size;
  const float* x    = (const float*)d_in[0];
  const float* ln_g = (const float*)d_in[1];
  const float* ln_b = (const float*)d_in[2];
  const float* Wq   = (const float*)d_in[3];
  const float* bq   = (const float*)d_in[4];
  const float* Wk   = (const float*)d_in[5];
  const float* bk   = (const float*)d_in[6];
  const float* Wv   = (const float*)d_in[7];
  const float* bv   = (const float*)d_in[8];
  const float* Wo   = (const float*)d_in[9];
  const float* bo   = (const float*)d_in[10];
  float* out = (float*)d_out;

  unsigned char* ws = (unsigned char*)d_ws;
  size_t off = 0;
  auto alloc = [&](size_t bytes) {
    unsigned char* p = ws + off;
    off += (bytes + 255) & ~(size_t)255;
    return p;
  };
  bf16* nx   = (bf16*)alloc((size_t)M_ * C_ * 2);
  bf16* Wqt  = (bf16*)alloc((size_t)C_ * C_ * 2);
  bf16* Wkt  = (bf16*)alloc((size_t)C_ * C_ * 2);
  bf16* Wvt  = (bf16*)alloc((size_t)C_ * C_ * 2);
  bf16* Wot  = (bf16*)alloc((size_t)C_ * C_ * 2);
  bf16* qb   = (bf16*)alloc((size_t)M_ * C_ * 2);
  bf16* kbuf = (bf16*)alloc((size_t)M_ * C_ * 2);
  bf16* vtb  = (bf16*)alloc((size_t)M_ * C_ * 2);
  bf16* ctx  = (bf16*)alloc((size_t)M_ * C_ * 2);

  ln_kernel<<<M_, 256, 0, stream>>>(x, ln_g, ln_b, nx);
  dim3 tg(C_ / 32, C_ / 32);
  wtrans_kernel<<<tg, 256, 0, stream>>>(Wq, Wqt);
  wtrans_kernel<<<tg, 256, 0, stream>>>(Wk, Wkt);
  wtrans_kernel<<<tg, 256, 0, stream>>>(Wv, Wvt);
  wtrans_kernel<<<tg, 256, 0, stream>>>(Wo, Wot);
  dim3 gg(C_ / 64, M_ / 128);
  gemm_kernel<<<gg, 256, 0, stream>>>(nx, Wqt, bq, qb,   nullptr, 0);
  gemm_kernel<<<gg, 256, 0, stream>>>(nx, Wkt, bk, kbuf, nullptr, 0);
  gemm_kernel<<<gg, 256, 0, stream>>>(nx, Wvt, bv, vtb,  nullptr, 1);
  rope_kernel<<<(B_ * H_ * T_ * 64) / 256, 256, 0, stream>>>(qb);
  rope_kernel<<<(B_ * H_ * T_ * 64) / 256, 256, 0, stream>>>(kbuf);
  attn_kernel<<<B_ * H_ * (T_ / 64), 128, 0, stream>>>(qb, kbuf, vtb, ctx);
  gemm_kernel<<<gg, 256, 0, stream>>>(ctx, Wot, bo, out, x, 2);
}
